// AreaAttention_77524159693071
// MI455X (gfx1250) — compile-verified, matrix operand order checked by reading, not run on hardware
//
#include <hip/hip_runtime.h>
#include <hip/hip_bf16.h>
#include <math.h>

// ---------------------------------------------------------------------------
// AreaAttention on gfx1250 (MI455X), all three GEMM stages on
// v_wmma_f32_16x16x32_f16, flash-attention with transposed-S register trick.
//
//  stage 0a: convert qkv_w / proj_w fp32 -> f16
//  stage 0b: transpose x [B,C,HW] fp32 -> xT f16 [B*HW][C]   (LDS tiled)
//  stage 1 : WMMA GEMM qkv = silu(bn(W_qkv @ x)); scatter to
//            Q f16 [b,h,s,d] (pre-scaled by 1/8), K f16 [b,h,s,d],
//            Vt f16 [b,h,d,s]
//  stage 2 : flash attention, 1 wave per (b,h,16-query tile);
//            S^T = K Q^T (2 tiles x 2 K-chunks), online softmax,
//            O^T += V^T P^T; writes Oattn f16 [b*HW][C]
//  stage 3 : WMMA GEMM out = silu(bn(W_proj @ Oattn)) fp32 [B,C,H,W]
// ---------------------------------------------------------------------------

typedef __attribute__((ext_vector_type(16))) _Float16 v16h;
typedef __attribute__((ext_vector_type(8)))  _Float16 v8h;
typedef __attribute__((ext_vector_type(8)))  float    v8f;

#define EPS 1e-5f

// Load a 16-bit WMMA A/B operand fragment for one lane: two contiguous
// 8-half (16B) chunks.  For K-base kb: lane<16 uses offsets {kb, kb+16},
// lane>=16 uses {kb+8, kb+24}  (ISA 16-bit A/B VGPR layout).
__device__ __forceinline__ v16h load_frag(const _Float16* rowp, int off0, int off1) {
  v8h a = *(const v8h*)(rowp + off0);
  v8h b = *(const v8h*)(rowp + off1);
  v16h r;
#pragma unroll
  for (int j = 0; j < 8; ++j) { r[j] = a[j]; r[8 + j] = b[j]; }
  return r;
}

__device__ __forceinline__ v8f wmma16(v16h a, v16h b, v8f c) {
  return __builtin_amdgcn_wmma_f32_16x16x32_f16(false, a, false, b, (short)0, c,
                                                false, false);
}

__device__ __forceinline__ float bn_silu(float y, float g, float be, float mn, float vr) {
  float sc = g * rsqrtf(vr + EPS);
  float v  = y * sc + (be - mn * sc);
  return v / (1.0f + __expf(-v));
}

// ---------------------------------------------------------------- stage 0a
__global__ void __launch_bounds__(256) k_wconv(const float* __restrict__ wq,
                                               const float* __restrict__ wp,
                                               _Float16* __restrict__ wq16,
                                               _Float16* __restrict__ wp16) {
  int i = blockIdx.x * 256 + threadIdx.x;              // 262144 threads total
  if (i < 768 * 256) wq16[i] = (_Float16)wq[i];
  int j = i - 768 * 256;
  if (j >= 0 && j < 256 * 256) wp16[j] = (_Float16)wp[j];
}

// ---------------------------------------------------------------- stage 0b
// x fp32 [B=4][C=256][HW=4096]  ->  xT f16 [col=B*HW][C]
__global__ void __launch_bounds__(256) k_xpose(const float* __restrict__ x,
                                               _Float16* __restrict__ xT) {
  __shared__ float t[16][17];
  int tx = threadIdx.x, ty = threadIdx.y;
  int col0 = blockIdx.x * 16;        // spatial tile (never crosses batch)
  int c0   = blockIdx.y * 16;        // channel tile
  int col  = col0 + tx;
  int b = col >> 12, s = col & 4095;
  t[ty][tx] = x[((size_t)(b * 256 + c0 + ty)) * 4096 + s];  // coalesced read
  __syncthreads();
  xT[(size_t)(col0 + ty) * 256 + c0 + tx] = (_Float16)t[tx][ty];
}

// ---------------------------------------------------------------- stage 1
// one wave -> 16 output channels x 16 columns, K=256 in 8 WMMA steps
__global__ void __launch_bounds__(256) k_qkv_gemm(
    const _Float16* __restrict__ w16, const _Float16* __restrict__ xT,
    const float* __restrict__ g, const float* __restrict__ be,
    const float* __restrict__ mn, const float* __restrict__ vr,
    _Float16* __restrict__ Q, _Float16* __restrict__ K,
    _Float16* __restrict__ Vt) {
  int w    = blockIdx.x * 8 + (threadIdx.x >> 5);
  int lane = threadIdx.x & 31;
  int lo = lane & 15, hi = lane >> 4;
  int off0 = hi * 8, off1 = 16 + hi * 8;
  int ob = (w >> 10) << 4;          // 48 o-tiles
  int cb = (w & 1023) << 4;         // 1024 col-tiles
  const _Float16* arow = w16 + (size_t)(ob + lo) * 256;
  const _Float16* brow = xT + (size_t)(cb + lo) * 256;
  v8f acc = {};
#pragma unroll
  for (int cc = 0; cc < 256; cc += 32)
    acc = wmma16(load_frag(arow, cc + off0, cc + off1),
                 load_frag(brow, cc + off0, cc + off1), acc);

  int col = cb + lo;
  int bb = col >> 12, s = col & 4095;
#pragma unroll
  for (int i = 0; i < 8; ++i) {
    int o = ob + i + hi * 8;                 // D-layout: row = i + 8*hi
    float y = bn_silu(acc[i], g[o], be[o], mn[o], vr[o]);
    int part = o >> 8, rem = o & 255, head = rem >> 6, d = rem & 63;
    size_t bh = (size_t)(bb * 4 + head);
    if (part == 0)
      Q[(bh * 4096 + s) * 64 + d] = (_Float16)(y * 0.125f);  // fold hd^-0.5
    else if (part == 1)
      K[(bh * 4096 + s) * 64 + d] = (_Float16)y;
    else
      Vt[(bh * 64 + d) * 4096 + s] = (_Float16)y;            // transposed
  }
}

// ---------------------------------------------------------------- stage 2
// flash attention: 1 wave per (b, head, 16-query tile); 4096 waves total
__global__ void __launch_bounds__(256) k_attn(const _Float16* __restrict__ Q,
                                              const _Float16* __restrict__ K,
                                              const _Float16* __restrict__ Vt,
                                              _Float16* __restrict__ O) {
  int w    = blockIdx.x * 8 + (threadIdx.x >> 5);
  int lane = threadIdx.x & 31;
  int lo = lane & 15, hi = lane >> 4;
  int off0 = hi * 8, off1 = 16 + hi * 8;
  int b = w >> 10, rem = w & 1023, h = rem >> 8, qb = (rem & 255) << 4;
  size_t bh = (size_t)(b * 4 + h);
  const _Float16* Qp = Q + bh * 4096 * 64;
  const _Float16* Kp = K + bh * 4096 * 64;
  const _Float16* Vp = Vt + bh * 64 * 4096;

  // Q^T B-operand, column q = qb+lo, K-dim = d (two 32-wide chunks)
  const _Float16* qrow = Qp + (size_t)(qb + lo) * 64;
  v16h bq0 = load_frag(qrow, off0, off1);
  v16h bq1 = load_frag(qrow, 32 + off0, 32 + off1);

  v8f acc0 = {}, acc1 = {}, acc2 = {}, acc3 = {};   // O^T: 4 d-tiles x 16 q
  float m = -1e30f, l = 0.0f;

  for (int kb = 0; kb < 4096; kb += 32) {
    // S^T tiles: rows = k, cols = q  (so softmax stats stay per-lane)
    const _Float16* k0row = Kp + (size_t)(kb + lo) * 64;
    const _Float16* k1row = Kp + (size_t)(kb + 16 + lo) * 64;
    v8f st0 = {}, st1 = {};
    st0 = wmma16(load_frag(k0row, off0, off1), bq0, st0);
    st0 = wmma16(load_frag(k0row, 32 + off0, 32 + off1), bq1, st0);
    st1 = wmma16(load_frag(k1row, off0, off1), bq0, st1);
    st1 = wmma16(load_frag(k1row, 32 + off0, 32 + off1), bq1, st1);

    float mx = st0[0];
#pragma unroll
    for (int i = 0; i < 8; ++i) {
      mx = fmaxf(mx, st0[i]);
      mx = fmaxf(mx, st1[i]);
    }
    mx = fmaxf(mx, __shfl_xor(mx, 16));        // join the two k-halves of q
    float mnew = fmaxf(m, mx);
    float corr = __expf(m - mnew);

    // exp in-place; D-layout of st0/st1 IS the B-operand layout of P^T
    float ps = 0.0f;
    v16h pb;
#pragma unroll
    for (int i = 0; i < 8; ++i) {
      float p0 = __expf(st0[i] - mnew);
      float p1 = __expf(st1[i] - mnew);
      ps += p0 + p1;
      pb[i]     = (_Float16)p0;
      pb[8 + i] = (_Float16)p1;
    }
    ps += __shfl_xor(ps, 16);
    l = l * corr + ps;
    m = mnew;
#pragma unroll
    for (int i = 0; i < 8; ++i) {
      acc0[i] *= corr; acc1[i] *= corr; acc2[i] *= corr; acc3[i] *= corr;
    }
    // O^T = V^T P^T : A rows = d (Vt is [d][s], s contiguous)
    acc0 = wmma16(load_frag(Vp + (size_t)(0 * 16 + lo) * 4096, kb + off0, kb + off1), pb, acc0);
    acc1 = wmma16(load_frag(Vp + (size_t)(1 * 16 + lo) * 4096, kb + off0, kb + off1), pb, acc1);
    acc2 = wmma16(load_frag(Vp + (size_t)(2 * 16 + lo) * 4096, kb + off0, kb + off1), pb, acc2);
    acc3 = wmma16(load_frag(Vp + (size_t)(3 * 16 + lo) * 4096, kb + off0, kb + off1), pb, acc3);
  }

  float inv = 1.0f / l;
  // O stored for stage-3 GEMM as [col = b*4096+q][c = h*64+d], d = 16t+i+8*hi
  _Float16* orow = O + ((size_t)(b * 4096 + qb + lo)) * 256 + h * 64 + hi * 8;
  v8h o0, o1, o2, o3;
#pragma unroll
  for (int i = 0; i < 8; ++i) {
    o0[i] = (_Float16)(acc0[i] * inv);
    o1[i] = (_Float16)(acc1[i] * inv);
    o2[i] = (_Float16)(acc2[i] * inv);
    o3[i] = (_Float16)(acc3[i] * inv);
  }
  *(v8h*)(orow + 0)  = o0;
  *(v8h*)(orow + 16) = o1;
  *(v8h*)(orow + 32) = o2;
  *(v8h*)(orow + 48) = o3;
}

// ---------------------------------------------------------------- stage 3
__global__ void __launch_bounds__(256) k_proj_gemm(
    const _Float16* __restrict__ w16, const _Float16* __restrict__ bT,
    const float* __restrict__ g, const float* __restrict__ be,
    const float* __restrict__ mn, const float* __restrict__ vr,
    float* __restrict__ out) {
  int w    = blockIdx.x * 8 + (threadIdx.x >> 5);
  int lane = threadIdx.x & 31;
  int lo = lane & 15, hi = lane >> 4;
  int off0 = hi * 8, off1 = 16 + hi * 8;
  int ob = (w >> 10) << 4;          // 16 o-tiles (C=256)
  int cb = (w & 1023) << 4;         // 1024 col-tiles
  const _Float16* arow = w16 + (size_t)(ob + lo) * 256;
  const _Float16* brow = bT + (size_t)(cb + lo) * 256;
  v8f acc = {};
#pragma unroll
  for (int cc = 0; cc < 256; cc += 32)
    acc = wmma16(load_frag(arow, cc + off0, cc + off1),
                 load_frag(brow, cc + off0, cc + off1), acc);

  int col = cb + lo;
  int bb = col >> 12, s = col & 4095;
#pragma unroll
  for (int i = 0; i < 8; ++i) {
    int o = ob + i + hi * 8;
    out[((size_t)(bb * 256 + o)) * 4096 + s] =
        bn_silu(acc[i], g[o], be[o], mn[o], vr[o]);
  }
}

// ---------------------------------------------------------------------------
extern "C" void kernel_launch(void* const* d_in, const int* in_sizes, int n_in,
                              void* d_out, int out_size, void* d_ws, size_t ws_size,
                              hipStream_t stream) {
  const float* x     = (const float*)d_in[0];
  const float* qkv_w = (const float*)d_in[1];
  const float* qg    = (const float*)d_in[2];
  const float* qb    = (const float*)d_in[3];
  const float* qm    = (const float*)d_in[4];
  const float* qv    = (const float*)d_in[5];
  const float* prj_w = (const float*)d_in[6];
  const float* pg    = (const float*)d_in[7];
  const float* pb    = (const float*)d_in[8];
  const float* pm    = (const float*)d_in[9];
  const float* pv    = (const float*)d_in[10];

  char* ws = (char*)d_ws;
  _Float16* wq16 = (_Float16*)(ws + 0);          //   768*256*2 = 384 KB
  _Float16* wp16 = (_Float16*)(ws + 393216);     //   256*256*2 = 128 KB
  _Float16* xT   = (_Float16*)(ws + 524288);     // 16384*256*2 =   8 MB
  _Float16* Qh   = (_Float16*)(ws + 8912896);    //               8 MB
  _Float16* Kh   = (_Float16*)(ws + 17301504);   //               8 MB
  _Float16* Vt   = (_Float16*)(ws + 25690112);   //               8 MB
  _Float16* Oh   = (_Float16*)(ws + 34078720);   //               8 MB  (end ~40.5 MB)

  k_wconv<<<1024, 256, 0, stream>>>(qkv_w, prj_w, wq16, wp16);
  k_xpose<<<dim3(1024, 16), dim3(16, 16), 0, stream>>>(x, xT);
  k_qkv_gemm<<<6144, 256, 0, stream>>>(wq16, xT, qg, qb, qm, qv, Qh, Kh, Vt);
  k_attn<<<512, 256, 0, stream>>>(Qh, Kh, Vt, Oh);
  k_proj_gemm<<<2048, 256, 0, stream>>>(wp16, Oh, pg, pb, pm, pv, (float*)d_out);
}
